// RelPosAttention_48163763258107
// MI455X (gfx1250) — matrix operand order repaired and checked by reading, not compiled, not run on hardware
//
#include <hip/hip_runtime.h>

// ---------------------------------------------------------------------------
// RelPosAttention (Swin window attention) for MI455X / gfx1250 (wave32, WMMA)
// B=128, N=256, C=512, heads=16, hd=32.
// All GEMMs run on v_wmma_f32_16x16x32_f16 (f16 in, f32 accumulate).
// Workspace (f16 activations, all L2-resident): ~174 MB.
// ---------------------------------------------------------------------------

typedef _Float16 v16h __attribute__((ext_vector_type(16)));
typedef _Float16 h8   __attribute__((ext_vector_type(8)));
typedef _Float16 h4   __attribute__((ext_vector_type(4)));
typedef float    v8f  __attribute__((ext_vector_type(8)));

typedef unsigned int u32x4 __attribute__((ext_vector_type(4)));
typedef int          i32x8 __attribute__((ext_vector_type(8)));
typedef int          i32x4 __attribute__((ext_vector_type(4)));

union HFrag { v16h v; h8 h[2]; };

__device__ __forceinline__ v8f wmma_f16(const v16h& a, const v16h& b, const v8f& c) {
  return __builtin_amdgcn_wmma_f32_16x16x32_f16(false, a, false, b, (short)0, c, false, false);
}

// --- TDM: 1D contiguous global -> LDS DMA via Tensor Data Mover -------------
// D# packed per CDNA5 ISA ch.8: group0 {count=1, lds_addr, global_addr, type=2},
// group1 {data_size=8B, tensor_dim0=tile_dim0=n8, dims1=1, stride0=n8}.
// This toolchain exposes the 6-arg builtin:
//   (uint32x4 g0, int32x8 g1, int32x4 g2, int32x4 g3, int32x8 pad, i32 cpol)
__device__ __forceinline__ void tdm_load_1d_to_lds(void* lds_ptr, const void* gptr,
                                                   unsigned bytes) {
  unsigned n8 = bytes >> 3;  // 8-byte elements (tile_dim0 is 16-bit: n8 <= 65535)
  unsigned long long ga = (unsigned long long)(size_t)gptr;
  unsigned lds = (unsigned)(size_t)lds_ptr;  // generic low 32 bits == LDS byte addr
  u32x4 g0 = { 1u,                                   // count=1, no gather
               lds,
               (unsigned)(ga & 0xFFFFFFFFu),
               (unsigned)((ga >> 32) & 0x01FFFFFFu) | 0x80000000u };  // type=2
  i32x8 g1 = { (int)(3u << 16),                       // data_size=3 (8 bytes)
               (int)((n8 & 0xFFFFu) << 16),           // tensor_dim0[15:0]
               (int)(((n8 >> 16) & 0xFFFFu) | (1u << 16)), // td0[31:16] | td1=1
               (int)((n8 & 0xFFFFu) << 16),           // td1 hi=0 | tile_dim0=n8
               1,                                     // tile_dim1=1, tile_dim2=0
               (int)n8,                               // tensor_dim0_stride[31:0]
               0, 0 };
  i32x4 z4 = { 0, 0, 0, 0 };
  i32x8 z8 = { 0, 0, 0, 0, 0, 0, 0, 0 };
  __builtin_amdgcn_tensor_load_to_lds(g0, g1, z4, z4, z8, 0);
}

// --- kernel 0: fp32 -> f16 convert (vec4) -----------------------------------
__global__ void k_cvt_f16(const float* __restrict__ src, _Float16* __restrict__ dst,
                          int n4) {
  int i = blockIdx.x * blockDim.x + threadIdx.x;
  if (i < n4) {
    float4 f = ((const float4*)src)[i];
    h4 h; h[0] = (_Float16)f.x; h[1] = (_Float16)f.y;
    h[2] = (_Float16)f.z; h[3] = (_Float16)f.w;
    ((h4*)dst)[i] = h;
  }
}

// --- kernel 0b: expand relative-position bias to [16][256][256] fp32 --------
__global__ void k_bias(const int* __restrict__ rel, const float* __restrict__ table,
                       float* __restrict__ biasf) {
  int i = blockIdx.x * 256 + threadIdx.x;  // 0..65535
  int idx = rel[i] * 16;
#pragma unroll
  for (int h = 0; h < 16; ++h) biasf[h * 65536 + i] = table[idx + h];
}

// --- kernel 1: QKV GEMM  [32768 x 512] x [512 x 1536]^T + bias --------------
// 8 waves/block; wave w -> 16 rows x 64 cols; q scaled, k row-major, v transposed.
__global__ __launch_bounds__(256) void k_qkv(const _Float16* __restrict__ xh,
                                             const _Float16* __restrict__ wh,
                                             const float* __restrict__ bias,
                                             _Float16* __restrict__ q,
                                             _Float16* __restrict__ k,
                                             _Float16* __restrict__ vT) {
  const int lane = threadIdx.x & 31, w = threadIdx.x >> 5;
  const int r = lane & 15, hh = lane >> 4;
  const int row0 = blockIdx.y * 128 + w * 16;
  const int col0 = blockIdx.x * 64;
  v8f acc[4] = {};
  const _Float16* arow = xh + (size_t)(row0 + r) * 512;
#pragma unroll 4
  for (int kk = 0; kk < 512; kk += 32) {
    __builtin_prefetch(arow + kk + 128, 0, 1);
    HFrag a;
    a.h[0] = *(const h8*)(arow + kk + hh * 16);        // K = hh*16 + 0..7
    a.h[1] = *(const h8*)(arow + kk + hh * 16 + 8);    // K = hh*16 + 8..15
#pragma unroll
    for (int c = 0; c < 4; ++c) {
      const _Float16* brow = wh + (size_t)(col0 + c * 16 + r) * 512 + kk + hh * 16;
      HFrag b;
      b.h[0] = *(const h8*)brow;
      b.h[1] = *(const h8*)(brow + 8);
      acc[c] = wmma_f16(a.v, b.v, acc[c]);
    }
  }
  const float scale = 0.17677669529663687f;  // 32^-0.5
#pragma unroll
  for (int c = 0; c < 4; ++c) {
    int o = col0 + c * 16 + r;
    float bo = bias[o];
    int sec = o >> 9, oc = o & 511, head = oc >> 5, d = oc & 31;
#pragma unroll
    for (int j = 0; j < 8; ++j) {
      int t = row0 + j + 8 * hh;        // D-layout: M = j + 8*(lane>=16)
      int bb = t >> 8, n = t & 255;
      size_t bh = (size_t)bb * 16 + head;
      float val = acc[c][j] + bo;
      if (sec == 0)      q[(bh * 256 + n) * 32 + d] = (_Float16)(val * scale);
      else if (sec == 1) k[(bh * 256 + n) * 32 + d] = (_Float16)val;
      else               vT[(bh * 32 + d) * 256 + n] = (_Float16)val;
    }
  }
}

// --- kernel 2: fused attention per (batch, head) -----------------------------
// 16 waves x 16 rows; S = q@k^T (K=hd=32, one WMMA per tile) + bias; register
// softmax with 16-lane shfl reductions; P through LDS; O = P@V via WMMA.
__global__ __launch_bounds__(512) void k_attn(const _Float16* __restrict__ q,
                                              const _Float16* __restrict__ kmat,
                                              const _Float16* __restrict__ vT,
                                              const float* __restrict__ biasf,
                                              _Float16* __restrict__ y) {
  __shared__ _Float16 kTile[256 * 32];   // 16 KB  [n][d]
  __shared__ _Float16 vTile[32 * 256];   // 16 KB  [d][m]
  __shared__ _Float16 pTile[256 * 256];  // 128 KB [n][m]
  const int bh = blockIdx.x, b = bh >> 4, head = bh & 15;
  const int lane = threadIdx.x & 31, w = threadIdx.x >> 5;
  const int r = lane & 15, hh = lane >> 4;
  const int row0 = w * 16;

  if (w == 0) {  // wave-uniform: TDM DMA of this head's K and V^T slabs
    tdm_load_1d_to_lds(kTile, kmat + (size_t)bh * 8192, 16384);
    tdm_load_1d_to_lds(vTile, vT + (size_t)bh * 8192, 16384);
    __builtin_amdgcn_s_wait_tensorcnt(0);
  }
  __syncthreads();

  HFrag qa;
  const _Float16* qrow = q + ((size_t)bh * 256 + row0 + r) * 32;
  qa.h[0] = *(const h8*)(qrow + hh * 16);
  qa.h[1] = *(const h8*)(qrow + hh * 16 + 8);

  v8f s[16];
#pragma unroll
  for (int ct = 0; ct < 16; ++ct) {
    HFrag kb;
    const _Float16* kr = &kTile[(ct * 16 + r) * 32 + hh * 16];
    kb.h[0] = *(const h8*)kr;
    kb.h[1] = *(const h8*)(kr + 8);
    v8f z = {};
    s[ct] = wmma_f16(qa.v, kb.v, z);
  }
  const float* bb = biasf + (size_t)head * 65536;
#pragma unroll
  for (int ct = 0; ct < 16; ++ct)
#pragma unroll
    for (int j = 0; j < 8; ++j)
      s[ct][j] += bb[(row0 + j + 8 * hh) * 256 + ct * 16 + r];

  // softmax over m (row lives in 16 lanes of one half + 16 col-tiles)
#pragma unroll
  for (int j = 0; j < 8; ++j) {
    float m = s[0][j];
#pragma unroll
    for (int ct = 1; ct < 16; ++ct) m = fmaxf(m, s[ct][j]);
    for (int d = 1; d < 16; d <<= 1) m = fmaxf(m, __shfl_xor(m, d, 32));
    float sum = 0.f;
#pragma unroll
    for (int ct = 0; ct < 16; ++ct) {
      float e = __expf(s[ct][j] - m);
      s[ct][j] = e;
      sum += e;
    }
    for (int d = 1; d < 16; d <<= 1) sum += __shfl_xor(sum, d, 32);
    float inv = 1.0f / sum;
#pragma unroll
    for (int ct = 0; ct < 16; ++ct)
      pTile[(row0 + j + 8 * hh) * 256 + ct * 16 + r] = (_Float16)(s[ct][j] * inv);
  }
  // in-wave LDS RAW (P write -> P read): compiler inserts s_wait_dscnt

  v8f oacc[2] = {};
#pragma unroll
  for (int ms = 0; ms < 8; ++ms) {
    HFrag pa;
    const _Float16* prow = &pTile[(row0 + r) * 256 + ms * 32];
    pa.h[0] = *(const h8*)(prow + hh * 16);
    pa.h[1] = *(const h8*)(prow + hh * 16 + 8);
#pragma unroll
    for (int c = 0; c < 2; ++c) {
      HFrag vb;
      const _Float16* vr = &vTile[(c * 16 + r) * 256 + ms * 32 + hh * 16];
      vb.h[0] = *(const h8*)vr;
      vb.h[1] = *(const h8*)(vr + 8);
      oacc[c] = wmma_f16(pa.v, vb.v, oacc[c]);
    }
  }
#pragma unroll
  for (int c = 0; c < 2; ++c)
#pragma unroll
    for (int j = 0; j < 8; ++j) {
      int n = row0 + j + 8 * hh, d = c * 16 + r;
      y[((size_t)b * 256 + n) * 512 + head * 32 + d] = (_Float16)oacc[c][j];
    }
}

// --- kernel 3: output projection  [32768 x 512] x [512 x 512]^T + bias ------
__global__ __launch_bounds__(256) void k_proj(const _Float16* __restrict__ yh,
                                              const _Float16* __restrict__ wh,
                                              const float* __restrict__ bias,
                                              float* __restrict__ out) {
  const int lane = threadIdx.x & 31, w = threadIdx.x >> 5;
  const int r = lane & 15, hh = lane >> 4;
  const int row0 = blockIdx.y * 128 + w * 16;
  const int col0 = blockIdx.x * 64;
  v8f acc[4] = {};
  const _Float16* arow = yh + (size_t)(row0 + r) * 512;
#pragma unroll 4
  for (int kk = 0; kk < 512; kk += 32) {
    __builtin_prefetch(arow + kk + 128, 0, 1);
    HFrag a;
    a.h[0] = *(const h8*)(arow + kk + hh * 16);
    a.h[1] = *(const h8*)(arow + kk + hh * 16 + 8);
#pragma unroll
    for (int c = 0; c < 4; ++c) {
      const _Float16* brow = wh + (size_t)(col0 + c * 16 + r) * 512 + kk + hh * 16;
      HFrag b;
      b.h[0] = *(const h8*)brow;
      b.h[1] = *(const h8*)(brow + 8);
      acc[c] = wmma_f16(a.v, b.v, acc[c]);
    }
  }
#pragma unroll
  for (int c = 0; c < 4; ++c) {
    int o = col0 + c * 16 + r;
    float bo = bias[o];
#pragma unroll
    for (int j = 0; j < 8; ++j)
      out[(size_t)(row0 + j + 8 * hh) * 512 + o] = acc[c][j] + bo;
  }
}

// ---------------------------------------------------------------------------
extern "C" void kernel_launch(void* const* d_in, const int* in_sizes, int n_in,
                              void* d_out, int out_size, void* d_ws, size_t ws_size,
                              hipStream_t stream) {
  const float* x          = (const float*)d_in[0];  // [128,256,512]
  const float* qkv_w      = (const float*)d_in[1];  // [1536,512]
  const float* qkv_b      = (const float*)d_in[2];  // [1536]
  const float* proj_w     = (const float*)d_in[3];  // [512,512]
  const float* proj_b     = (const float*)d_in[4];  // [512]
  const float* bias_table = (const float*)d_in[5];  // [961,16]
  const int*   rel_index  = (const int*)d_in[6];    // [65536]

  char* ws = (char*)d_ws;
  _Float16* xh     = (_Float16*)ws; ws += (size_t)16777216 * 2;  // x f16
  _Float16* wqkvh  = (_Float16*)ws; ws += (size_t)786432 * 2;    // qkv_w f16
  _Float16* wprojh = (_Float16*)ws; ws += (size_t)262144 * 2;    // proj_w f16
  _Float16* qb     = (_Float16*)ws; ws += (size_t)16777216 * 2;  // q*scale [bh,n,d]
  _Float16* kb     = (_Float16*)ws; ws += (size_t)16777216 * 2;  // k [bh,n,d]
  _Float16* vTb    = (_Float16*)ws; ws += (size_t)16777216 * 2;  // v^T [bh,d,n]
  _Float16* yb     = (_Float16*)ws; ws += (size_t)16777216 * 2;  // pre-proj [b,n,c]
  float*    biasf  = (float*)ws;                                  // [16,256,256]

  k_cvt_f16<<<16384, 256, 0, stream>>>(x, xh, 4194304);
  k_cvt_f16<<<768, 256, 0, stream>>>(qkv_w, wqkvh, 196608);
  k_cvt_f16<<<256, 256, 0, stream>>>(proj_w, wprojh, 65536);
  k_bias<<<256, 256, 0, stream>>>(rel_index, bias_table, biasf);
  k_qkv<<<dim3(24, 256), 256, 0, stream>>>(xh, wqkvh, qkv_b, qb, kb, vTb);
  k_attn<<<2048, 512, 0, stream>>>(qb, kb, vTb, biasf, yb);
  k_proj<<<dim3(8, 256), 256, 0, stream>>>(yb, wprojh, proj_b, (float*)d_out);
}
